// TMSA_12876311953824
// MI455X (gfx1250) — matrix operand order repaired and checked by reading, not compiled, not run on hardware
//
#include <hip/hip_runtime.h>
#include <hip/hip_bf16.h>
#include <cstdint>
#include <cstddef>

// ---------------------------------------------------------------------------
// Types matching gfx1250 WMMA builtin signatures
// ---------------------------------------------------------------------------
typedef __bf16 bf16_t;
typedef __attribute__((ext_vector_type(16))) __bf16 v16bf;
typedef __attribute__((ext_vector_type(8)))  __bf16 v8bf;
typedef __attribute__((ext_vector_type(8)))  float  v8f;

#define NTOK 64      // tokens per window
#define CDIM 512     // model dim
#define NH   8       // heads
#define HD   64      // head dim
#define NWIN 2048    // windows
#define QKVC 1536    // 3*C

__device__ __forceinline__ bf16_t f2bf(float f) {
  uint32_t u = __builtin_bit_cast(uint32_t, f);
  u += 0x7FFFu + ((u >> 16) & 1u);           // round-to-nearest-even
  uint16_t h = (uint16_t)(u >> 16);
  return __builtin_bit_cast(bf16_t, h);
}

__device__ __forceinline__ v8f zero8() {
  v8f v;
#pragma unroll
  for (int i = 0; i < 8; ++i) v[i] = 0.f;
  return v;
}

__device__ __forceinline__ v8f wmma_bf16(v16bf a, v16bf b, v8f c) {
  // 8 args: (neg_a, A, neg_b, B, c_mod, C, reuse_a, reuse_b)
  return __builtin_amdgcn_wmma_f32_16x16x32_bf16(false, a, false, b, (short)0, c,
                                                 false, false);
}

// A-fragment: 16x32 bf16 tile from row-major storage (stride `ld` elements).
// Layout: lanes 0-15 row M=lane, K 0..15; lanes 16-31 same rows,
// K 16..31  -> two contiguous 16B chunks per lane.
__device__ __forceinline__ v16bf load_a(const bf16_t* base, int ld, int row0, int col0) {
  const int lane = threadIdx.x & 31;
  const bf16_t* p = base + (size_t)(row0 + (lane & 15)) * ld + col0 + ((lane >> 4) << 4);
  v8bf lo = *(const v8bf*)p;
  v8bf hi = *(const v8bf*)(p + 8);
  return __builtin_shufflevector(lo, hi, 0,1,2,3,4,5,6,7,8,9,10,11,12,13,14,15);
}

// B-fragment: 32x16 tile from packed layout pk[kt][N][g(2)][e(16)]
// where element = B[kt*32 + g*16 + e][n].  Per lane: one contiguous 32B read.
__device__ __forceinline__ v16bf load_b(const bf16_t* pk, int N, int kt, int n0) {
  const int lane = threadIdx.x & 31;
  const bf16_t* p = pk + (((size_t)(kt * N + n0 + (lane & 15))) << 5) + ((lane >> 4) << 4);
  return *(const v16bf*)p;
}

// ---------------------------------------------------------------------------
// Prep kernel 1: repack an fp32 KxN weight into bf16 B-fragment-friendly
// layout [K/32][N][2][16].
// ---------------------------------------------------------------------------
__global__ void pack_b_kernel(const float* __restrict__ w, bf16_t* __restrict__ dst,
                              int N, int total) {
  int i = blockIdx.x * blockDim.x + threadIdx.x;
  if (i >= total) return;
  int k = i / N, n = i - k * N;
  dst[(((size_t)((k >> 5) * N + n)) << 5) + (((k >> 4) & 1) << 4) + (k & 15)] = f2bf(w[i]);
}

// ---------------------------------------------------------------------------
// Prep kernel 2: tb[b][j] = b_qkv[j] + temb[b,:] @ w_qkv_t[:,j]   (fp32)
// One block per window; temb row staged in LDS; consecutive threads read
// consecutive columns of w_qkv_t (coalesced).
// ---------------------------------------------------------------------------
__global__ void temb_bias_kernel(const float* __restrict__ temb,
                                 const float* __restrict__ w_qkv_t,
                                 const float* __restrict__ b_qkv,
                                 float* __restrict__ tb) {
  __shared__ float te[CDIM];
  const int b = blockIdx.x;
  for (int i = threadIdx.x; i < CDIM; i += blockDim.x) te[i] = temb[(size_t)b * CDIM + i];
  __syncthreads();
  for (int j = threadIdx.x; j < QKVC; j += blockDim.x) {
    float s = b_qkv[j];
#pragma unroll 4
    for (int k = 0; k < CDIM; ++k) s += te[k] * w_qkv_t[(size_t)k * QKVC + j];
    tb[(size_t)b * QKVC + j] = s;
  }
}

// ---------------------------------------------------------------------------
// Fused windowed-attention kernel.
// 1 block = 1 window, 256 threads = 8 waves; wave w handles head w in the
// attention stages.  Dynamic LDS layout (256 KB total):
//   [0      .. 64K)  xs : x bf16 [64][512]      -> reused as P [8][64][64]
//   [64K    ..128K)  qO : q bf16 [8][64][64]    -> reused as O [64][512]
//   [128K   ..192K)  kpk: k^T packed [8][2][64][2][16]   (B layout, K=d)
//   [192K   ..256K)  vpk: v packed   [8][2][64][2][16]   (B layout, K=token)
// ---------------------------------------------------------------------------
__global__ void __launch_bounds__(256)
fused_window_attn(const float* __restrict__ x,
                  const float* __restrict__ rpb,       // (225, 8)
                  const float* __restrict__ b_proj,
                  const bf16_t* __restrict__ wqkv_pk,  // packed (512,1536)
                  const bf16_t* __restrict__ wproj_pk, // packed (512,512)
                  const float* __restrict__ tb,        // (B,1536)
                  float* __restrict__ out) {
  extern __shared__ char smem[];
  bf16_t* xs  = (bf16_t*)(smem);
  bf16_t* qO  = (bf16_t*)(smem + 65536);
  bf16_t* kpk = (bf16_t*)(smem + 131072);
  bf16_t* vpk = (bf16_t*)(smem + 196608);

  const int b    = blockIdx.x;
  const int w    = threadIdx.x >> 5;
  const int lane = threadIdx.x & 31;
  const int hi   = lane >> 4;
  const int lo   = lane & 15;

  // ---- Stage 1: stage x window in LDS as bf16 ----
  {
    const float4* xb = (const float4*)(x + (size_t)b * NTOK * CDIM);
    for (int i = threadIdx.x; i < NTOK * CDIM / 4; i += 256) {
      float4 f = xb[i];
      bf16_t* d = xs + (size_t)i * 4;
      d[0] = f2bf(f.x); d[1] = f2bf(f.y); d[2] = f2bf(f.z); d[3] = f2bf(f.w);
    }
  }
  __syncthreads();

  // ---- Stage 2: qkv = x @ w_qkv + tb  ->  q/kT/v into LDS ----
  {
    const float* tbb = tb + (size_t)b * QKVC;
    for (int t = 0; t < 12; ++t) {
      const int n0 = w * 192 + t * 16;
      v8f acc[4] = {zero8(), zero8(), zero8(), zero8()};
      for (int kt = 0; kt < 16; ++kt) {
        if (kt + 1 < 16)  // pull next K-slice of weights from L2
          __builtin_prefetch(wqkv_pk + (((size_t)((kt + 1) * QKVC + n0)) << 5), 0, 1);
        v16bf bfrag = load_b(wqkv_pk, QKVC, kt, n0);
#pragma unroll
        for (int mt = 0; mt < 4; ++mt) {
          v16bf afrag = load_a(xs, CDIM, mt * 16, kt * 32);
          acc[mt] = wmma_bf16(afrag, bfrag, acc[mt]);
        }
      }
      const int   n    = n0 + lo;
      const float bias = tbb[n];
      const int   part = n >> 9;     // 0=q 1=k 2=v (uniform across tile)
      const int   cn   = n & 511;
      const int   h    = cn >> 6;
      const int   dc   = cn & 63;
#pragma unroll
      for (int mt = 0; mt < 4; ++mt) {
        const int mb = mt * 16 + hi * 8;
#pragma unroll
        for (int r = 0; r < 8; ++r) {
          const float val = acc[mt][r] + bias;
          const int m = mb + r;
          if (part == 0) {
            qO[(h * 64 + m) * 64 + dc] = f2bf(val * 0.125f);          // q * D^-0.5
          } else if (part == 1) {
            // k^T packed: K-dim = dc, n = token m
            kpk[((h * 2 + (dc >> 5)) * 64 + m) * 32 + (((dc >> 4) & 1) << 4) + (dc & 15)] = f2bf(val);
          } else {
            // v packed: K-dim = token m, n = dc
            vpk[((h * 2 + (m >> 5)) * 64 + dc) * 32 + (((m >> 4) & 1) << 4) + (m & 15)] = f2bf(val);
          }
        }
      }
    }
  }
  __syncthreads();

  // ---- Stage 3: per-head attention (wave w == head w) ----
  {
    const int h = w;
    const bf16_t* qh = qO  + h * 64 * 64;
    const bf16_t* kh = kpk + h * 2 * 64 * 32;
    const bf16_t* vh = vpk + h * 2 * 64 * 32;
    bf16_t*       Ph = xs  + h * 64 * 64;     // reuse x region for probs

    // S = q @ k^T  + rel-pos bias, softmax -> P (bf16)
    for (int mt = 0; mt < 4; ++mt) {
      v8f s[4] = {zero8(), zero8(), zero8(), zero8()};
#pragma unroll
      for (int kt = 0; kt < 2; ++kt) {
        v16bf a = load_a(qh, 64, mt * 16, kt * 32);
#pragma unroll
        for (int nt = 0; nt < 4; ++nt)
          s[nt] = wmma_bf16(a, load_b(kh, 64, kt, nt * 16), s[nt]);
      }
#pragma unroll
      for (int nt = 0; nt < 4; ++nt) {
        const int n  = nt * 16 + lo;
        const int jy = n >> 3, jx = n & 7;
#pragma unroll
        for (int r = 0; r < 8; ++r) {
          const int m  = mt * 16 + hi * 8 + r;
          const int iy = m >> 3, ix = m & 7;
          const int idx = (iy - jy + 7) * 15 + (ix - jx + 7);
          s[nt][r] += rpb[idx * NH + h];
        }
      }
      // row softmax: each row lives in one 16-lane half (cols across lanes,
      // 4 n-tiles in registers)
#pragma unroll
      for (int r = 0; r < 8; ++r) {
        float mx = fmaxf(fmaxf(s[0][r], s[1][r]), fmaxf(s[2][r], s[3][r]));
#pragma unroll
        for (int msk = 8; msk >= 1; msk >>= 1) mx = fmaxf(mx, __shfl_xor(mx, msk, 32));
        const float e0 = __expf(s[0][r] - mx);
        const float e1 = __expf(s[1][r] - mx);
        const float e2 = __expf(s[2][r] - mx);
        const float e3 = __expf(s[3][r] - mx);
        float sm = e0 + e1 + e2 + e3;
#pragma unroll
        for (int msk = 8; msk >= 1; msk >>= 1) sm += __shfl_xor(sm, msk, 32);
        const float inv = 1.0f / sm;
        const int m = mt * 16 + hi * 8 + r;
        Ph[m * 64 +  0 + lo] = f2bf(e0 * inv);
        Ph[m * 64 + 16 + lo] = f2bf(e1 * inv);
        Ph[m * 64 + 32 + lo] = f2bf(e2 * inv);
        Ph[m * 64 + 48 + lo] = f2bf(e3 * inv);
      }
    }
    __syncthreads();   // all waves done reading q, P fully written

    // O = P @ v  -> store bf16 into qO reinterpreted as O[64][512]
    for (int mt = 0; mt < 4; ++mt) {
      v8f o[4] = {zero8(), zero8(), zero8(), zero8()};
#pragma unroll
      for (int kt = 0; kt < 2; ++kt) {
        v16bf a = load_a(Ph, 64, mt * 16, kt * 32);
#pragma unroll
        for (int nt = 0; nt < 4; ++nt)
          o[nt] = wmma_bf16(a, load_b(vh, 64, kt, nt * 16), o[nt]);
      }
#pragma unroll
      for (int nt = 0; nt < 4; ++nt)
#pragma unroll
        for (int r = 0; r < 8; ++r) {
          const int m = mt * 16 + hi * 8 + r;
          qO[m * CDIM + h * 64 + nt * 16 + lo] = f2bf(o[nt][r]);
        }
    }
  }
  __syncthreads();

  // ---- Stage 4: out = O @ w_proj + b_proj -> global fp32 ----
  {
    const bf16_t* O = qO;
    for (int t = 0; t < 4; ++t) {
      const int n0 = w * 64 + t * 16;
      v8f acc[4] = {zero8(), zero8(), zero8(), zero8()};
      for (int kt = 0; kt < 16; ++kt) {
        v16bf bfrag = load_b(wproj_pk, CDIM, kt, n0);
#pragma unroll
        for (int mt = 0; mt < 4; ++mt) {
          v16bf a = load_a(O, CDIM, mt * 16, kt * 32);
          acc[mt] = wmma_bf16(a, bfrag, acc[mt]);
        }
      }
      const float bias = b_proj[n0 + lo];
#pragma unroll
      for (int mt = 0; mt < 4; ++mt)
#pragma unroll
        for (int r = 0; r < 8; ++r) {
          const int m = mt * 16 + hi * 8 + r;
          out[((size_t)b * NTOK + m) * CDIM + n0 + lo] = acc[mt][r] + bias;
        }
    }
  }
}

// ---------------------------------------------------------------------------
// Launch
// ---------------------------------------------------------------------------
extern "C" void kernel_launch(void* const* d_in, const int* in_sizes, int n_in,
                              void* d_out, int out_size, void* d_ws, size_t ws_size,
                              hipStream_t stream) {
  (void)in_sizes; (void)n_in; (void)out_size; (void)ws_size;
  const float* x       = (const float*)d_in[0];
  const float* temb    = (const float*)d_in[1];
  const float* w_qkv   = (const float*)d_in[2];
  const float* b_qkv   = (const float*)d_in[3];
  const float* w_qkv_t = (const float*)d_in[4];
  const float* w_proj  = (const float*)d_in[5];
  const float* b_proj  = (const float*)d_in[6];
  const float* rpb     = (const float*)d_in[7];

  // workspace layout: packed bf16 weights (L2-resident) + per-window qkv bias
  char*  ws       = (char*)d_ws;
  bf16_t* wqkv_pk  = (bf16_t*)ws;                                   // 1.5 MB
  bf16_t* wproj_pk = (bf16_t*)(ws + (size_t)CDIM * QKVC * 2);       // 0.5 MB
  float*  tb       = (float*)(ws + (size_t)CDIM * QKVC * 2
                                 + (size_t)CDIM * CDIM * 2);        // 12.6 MB

  pack_b_kernel<<<(CDIM * QKVC + 255) / 256, 256, 0, stream>>>(w_qkv, wqkv_pk, QKVC, CDIM * QKVC);
  pack_b_kernel<<<(CDIM * CDIM + 255) / 256, 256, 0, stream>>>(w_proj, wproj_pk, CDIM, CDIM * CDIM);
  temb_bias_kernel<<<NWIN, 256, 0, stream>>>(temb, w_qkv_t, b_qkv, tb);

  (void)hipFuncSetAttribute((const void*)fused_window_attn,
                            hipFuncAttributeMaxDynamicSharedMemorySize, 262144);
  fused_window_attn<<<NWIN, 256, 262144, stream>>>(x, rpb, b_proj, wqkv_pk, wproj_pk,
                                                   tb, (float*)d_out);
}
